// XFeat_69252052680989
// MI455X (gfx1250) — matrix operand order repaired and checked by reading, not compiled, not run on hardware
//
#include <hip/hip_runtime.h>
#include <cstdint>

// ---------------------------------------------------------------------------
// XFeat detect() for MI455X (gfx1250, wave32).
// Pipeline:
//   1. k_zero_border     : zero 4-px border of padded heatmap
//   2. k_softmax_shuffle : softmax(65) + 8x8 pixel shuffle -> padded K1h
//                          (float4 stores, 16B-aligned thanks to 4-px pad)
//   3. k_nms_score       : TDM tile->LDS, 5x5 NMS, bilinear(H1); positives
//                          pushed to per-batch sparse list + 4096-bin histogram
//                          in the same pass (no dense score plane)
//   4. k_cutoff/k_compact: histogram select of top-4096 candidates from the
//                          sparse positives list
//   5. k_sort            : per-batch 4096-wide bitonic sort (LDS), emits
//                          mkpts / scores / valid to d_out
//   6. k_norm_tr         : L2-normalize M1 over channels + transpose to NHWC
//                          (float4 loads/stores)
//   7. k_feats           : bicubic feature gather; per-16-keypoint L2 norms
//                          via v_wmma_f32_16x16x32_f16 (diag of F*F^T)
// Workspace requirement: ~287 MB.
// ---------------------------------------------------------------------------

typedef __attribute__((ext_vector_type(16))) _Float16 v16h;
typedef __attribute__((ext_vector_type(8)))  float    v8f;
typedef __attribute__((ext_vector_type(4)))  uint32_t u32x4;
typedef __attribute__((ext_vector_type(8)))  int32_t  i32x8;
typedef __attribute__((ext_vector_type(4)))  int32_t  i32x4;

#define B_     16
#define HC     136
#define WC     240
#define H_     1088
#define W_     1920
#define PAD    4             // border pad so interior is 16B-aligned
#define WP     1928          // padded pitch  (W_ + 2*PAD), WP*4 % 16 == 0
#define HP     1096          // padded height (H_ + 2*PAD)
#define HW     (H_ * W_)
#define HCWC   (HC * WC)
#define TOPK   4096
#define NBIN   4096
#define MAXPOS 131072        // per-batch positives capacity (>= worst case /16)
#define THRESH 0.05f
#define ACUB   (-0.75f)

// -------------------------- bicubic weights --------------------------------
__device__ __forceinline__ float cub1(float x) {
  return ((ACUB + 2.0f) * x - (ACUB + 3.0f)) * x * x + 1.0f;
}
__device__ __forceinline__ float cub2(float x) {
  return ((ACUB * x - 5.0f * ACUB) * x + 8.0f * ACUB) * x - 4.0f * ACUB;
}

// -------------------- 1. zero padded-heatmap border ------------------------
__global__ void k_zero_border(float* __restrict__ k1hp) {
  const int per = 8 * WP + 8 * H_;  // 24128 border cells per batch
  int i = blockIdx.x * blockDim.x + threadIdx.x;
  if (i >= B_ * per) return;
  int b = i / per, r = i % per;
  int y, x;
  if (r < 4 * WP)            { y = r / WP;               x = r % WP; }
  else if (r < 8 * WP)       { int q = r - 4 * WP; y = HP - 4 + q / WP; x = q % WP; }
  else {
    int q = r - 8 * WP; int row = q >> 3; int c = q & 7;
    y = PAD + row; x = (c < 4) ? c : (WP - 4 + (c - 4));
  }
  k1hp[((size_t)b * HP + y) * WP + x] = 0.0f;
}

// ------------------ 2. softmax over 65 bins + pixel shuffle ----------------
__global__ void k_softmax_shuffle(const float* __restrict__ K1,
                                  float* __restrict__ k1hp) {
  int cell = blockIdx.x * blockDim.x + threadIdx.x;
  if (cell >= B_ * HCWC) return;
  int b = cell / HCWC, p = cell % HCWC;
  int yc = p / WC, xc = p % WC;
  const float* src = K1 + (size_t)b * 65 * HCWC + p;
  float v[65];
  float mx = -3.4e38f;
#pragma unroll
  for (int c = 0; c < 65; ++c) { v[c] = src[(size_t)c * HCWC]; mx = fmaxf(mx, v[c]); }
  float sum = 0.0f;
#pragma unroll
  for (int c = 0; c < 65; ++c) { v[c] = __expf(v[c] - mx); sum += v[c]; }
  float inv = 1.0f / sum;
  // interior pixel (y,x) -> k1hp[(y+PAD)*WP + (x+PAD)], 16B-aligned rows of 8
  float* dst = k1hp + ((size_t)b * HP + (size_t)yc * 8 + PAD) * WP + (size_t)xc * 8 + PAD;
#pragma unroll
  for (int dy = 0; dy < 8; ++dy) {
    const int c0 = dy * 8;
    float4 lo = make_float4(v[c0 + 0] * inv, v[c0 + 1] * inv, v[c0 + 2] * inv, v[c0 + 3] * inv);
    float4 hi = make_float4(v[c0 + 4] * inv, v[c0 + 5] * inv, v[c0 + 6] * inv, v[c0 + 7] * inv);
    *(float4*)(dst + (size_t)dy * WP)     = lo;
    *(float4*)(dst + (size_t)dy * WP + 4) = hi;
  }
}

// ---- 3. NMS (5x5) + score: TDM tile staging, sparse positives + histogram --
__global__ void k_nms_score(const float* __restrict__ k1hp,
                            const float* __restrict__ H1,
                            float* __restrict__ posS, int* __restrict__ posI,
                            int* __restrict__ posCnt, int* __restrict__ hist) {
  __shared__ float sT[12 * 36];  // 32x8 tile + 2-px apron
  const int b = blockIdx.z, gy0 = blockIdx.y * 8, gx0 = blockIdx.x * 32;
  // apron origin: interior (gy0-2, gx0-2) -> padded (gy0+PAD-2, gx0+PAD-2)
  const float* tile0 = k1hp + ((size_t)b * HP + gy0 + PAD - 2) * WP + gx0 + PAD - 2;

#if __has_builtin(__builtin_amdgcn_tensor_load_to_lds) && __has_builtin(__builtin_amdgcn_s_wait_tensorcnt)
  if (threadIdx.x < 32) {  // single wave issues the TDM descriptor
    uint64_t ga = (uint64_t)(uintptr_t)tile0;
    uint32_t ldsOff = (uint32_t)(uintptr_t)(&sT[0]);  // low 32 bits = LDS offset
    u32x4 g0;
    g0.x = 1u;                                              // count=1, user mode
    g0.y = ldsOff;                                          // lds_addr
    g0.z = (uint32_t)ga;                                    // global_addr[31:0]
    g0.w = (uint32_t)((ga >> 32) & 0x01FFFFFFu) | (2u << 30);  // addr[56:32] | type=2
    const uint32_t dim0 = 0x100000u, dim1 = 0x100000u;      // huge dims: tile always in-bounds
    const uint32_t td0 = 36u, td1 = 12u, st0 = (uint32_t)WP;
    i32x8 g1;
    g1.s0 = (int)0x00020000u;                               // data_size = 4B
    g1.s1 = (int)((dim0 & 0xFFFFu) << 16);                  // tensor_dim0[15:0]
    g1.s2 = (int)((dim0 >> 16) | ((dim1 & 0xFFFFu) << 16)); // dim0 hi | dim1 lo
    g1.s3 = (int)((dim1 >> 16) | (td0 << 16));              // dim1 hi | tile_dim0
    g1.s4 = (int)(td1);                                     // tile_dim1 | tile_dim2=0
    g1.s5 = (int)st0;                                       // tensor_dim0_stride lo
    g1.s6 = 0; g1.s7 = 0;
    i32x4 z4 = {0, 0, 0, 0};
#if defined(__clang_major__) && (__clang_major__ >= 23)
    i32x8 z8 = {0, 0, 0, 0, 0, 0, 0, 0};
    __builtin_amdgcn_tensor_load_to_lds(g0, g1, z4, z4, z8, 0);
#else
    __builtin_amdgcn_tensor_load_to_lds(g0, g1, z4, z4, 0);
#endif
    __builtin_amdgcn_s_wait_tensorcnt(0);
  }
#else
  for (int i = threadIdx.x; i < 12 * 36; i += blockDim.x) {
    int ty = i / 36, tx = i % 36;
    sT[i] = tile0[(size_t)ty * WP + tx];
  }
#endif
  __syncthreads();

  int tx = threadIdx.x & 31, ty = threadIdx.x >> 5;  // 32x8 pixels, 256 threads
  int x = gx0 + tx, y = gy0 + ty;
  float c = sT[(ty + 2) * 36 + tx + 2];
  float m = -1.0f;
#pragma unroll
  for (int dy = 0; dy < 5; ++dy)
#pragma unroll
    for (int dx = 0; dx < 5; ++dx)
      m = fmaxf(m, sT[(ty + dy) * 36 + tx + dx]);
  bool pos = (c == m) && (c > THRESH);
  if (x == 0 && y == 0) pos = false;  // torch quirk: index 0 forced to -1

  if (pos) {
    // bilinear sample of H1 (coarse grid) at full-res pixel coords
    float ux = (float)x * (240.0f / 1919.0f) - 0.5f;
    float uy = (float)y * (136.0f / 1087.0f) - 0.5f;
    float xf = floorf(ux), yf = floorf(uy);
    float fx = ux - xf, fy = uy - yf;
    int x0 = (int)xf, y0 = (int)yf;
    const float* h1 = H1 + (size_t)b * HCWC;
    auto g = [&](int iy, int ix) -> float {
      return (iy >= 0 && iy < HC && ix >= 0 && ix < WC) ? h1[iy * WC + ix] : 0.0f;
    };
    float bil = (g(y0, x0) * (1.0f - fx) + g(y0, x0 + 1) * fx) * (1.0f - fy)
              + (g(y0 + 1, x0) * (1.0f - fx) + g(y0 + 1, x0 + 1) * fx) * fy;
    // nearest(K1h) at these coords is exactly the center pixel (shown analytically)
    float s = c * bil;
    if (s > 0.0f) {
      int bin = (int)(s * (float)NBIN);
      if (bin > NBIN - 1) bin = NBIN - 1;
      atomicAdd(&hist[b * NBIN + bin], 1);
      int p = atomicAdd(&posCnt[b], 1);
      if (p < MAXPOS) {
        posS[(size_t)b * MAXPOS + p] = s;
        posI[(size_t)b * MAXPOS + p] = y * W_ + x;
      }
    }
  }
}

// --------------------- 4a. per-batch cutoff bin ----------------------------
__global__ void k_cutoff(const int* __restrict__ hist, int* __restrict__ cutinfo) {
  int b = blockIdx.x;
  if (threadIdx.x == 0) {
    int cum = 0, cut = -1, nA = 0;
    for (int bin = NBIN - 1; bin >= 0; --bin) {
      int h = hist[b * NBIN + bin];
      cum += h;
      if (cum >= TOPK) { cut = bin; nA = cum - h; break; }
    }
    cutinfo[b * 2] = cut; cutinfo[b * 2 + 1] = nA;
  }
}

// ----------- 4b. compact candidates from the sparse positives list ---------
__global__ void k_compact(const float* __restrict__ posS, const int* __restrict__ posI,
                          const int* __restrict__ posCnt,
                          const int* __restrict__ cutinfo,
                          int* __restrict__ cnts,
                          float* __restrict__ candS, int* __restrict__ candI) {
  int i = blockIdx.x * blockDim.x + threadIdx.x;
  int b = i / MAXPOS, j = i % MAXPOS;
  if (b >= B_) return;
  int n = posCnt[b]; if (n > MAXPOS) n = MAXPOS;
  if (j >= n) return;
  float s = posS[(size_t)b * MAXPOS + j];
  int id = posI[(size_t)b * MAXPOS + j];
  int bin = (int)(s * (float)NBIN);
  if (bin > NBIN - 1) bin = NBIN - 1;
  int cut = cutinfo[b * 2], nA = cutinfo[b * 2 + 1];
  if (bin > cut) {
    int p = atomicAdd(&cnts[b * 2], 1);
    if (p < TOPK) { candS[b * TOPK + p] = s; candI[b * TOPK + p] = id; }
  } else if (bin == cut) {
    int p = nA + atomicAdd(&cnts[b * 2 + 1], 1);
    if (p < TOPK) { candS[b * TOPK + p] = s; candI[b * TOPK + p] = id; }
  }
}

// --------------------- 5. per-batch bitonic sort (desc) --------------------
__global__ void __launch_bounds__(512)
k_sort(const float* __restrict__ candS, const int* __restrict__ candI,
       int* __restrict__ topI, float* __restrict__ out) {
  __shared__ float ss[TOPK];
  __shared__ int   si[TOPK];
  int b = blockIdx.x, t = threadIdx.x;
  for (int i = t; i < TOPK; i += 512) { ss[i] = candS[b * TOPK + i]; si[i] = candI[b * TOPK + i]; }
  __syncthreads();
  for (int k = 2; k <= TOPK; k <<= 1) {
    for (int j = k >> 1; j > 0; j >>= 1) {
      for (int pos = t; pos < TOPK; pos += 512) {
        int q = pos ^ j;
        if (q > pos) {
          float a = ss[pos], c = ss[q];
          int ia = si[pos], ic = si[q];
          bool up = ((pos & k) == 0);                     // final pass: descending
          bool good = (a > c) || (a == c && ia <= ic);    // tie-break: lower index first
          if (up ? !good : good) { ss[pos] = c; ss[q] = a; si[pos] = ic; si[q] = ia; }
        }
      }
      __syncthreads();
    }
  }
  float* mk = out;
  float* st = out + (size_t)B_ * TOPK * 2;
  float* va = out + (size_t)B_ * TOPK * 67;  // after mkpts + scores + feats
  for (int i = t; i < TOPK; i += 512) {
    float s = ss[i]; int id = si[i];
    float so = (s > 0.0f) ? s : -1.0f;       // sentinel slots (score==0) -> -1
    int kx = id % W_, ky = id / W_;
    size_t o = (size_t)b * TOPK + i;
    mk[o * 2 + 0] = (float)kx;
    mk[o * 2 + 1] = (float)ky;
    st[o] = so;
    va[o] = (so > 0.0f) ? 1.0f : 0.0f;
    topI[b * TOPK + i] = id;
  }
}

// ---------------- 6. L2-normalize M1 + transpose to NHWC -------------------
__global__ void k_norm_tr(const float* __restrict__ M1, float* __restrict__ m1n) {
  __shared__ float ld[64][65];
  __shared__ float sc[64];
  int blk = blockIdx.x;
  int b = blk / 510, p0 = (blk % 510) * 64;   // HCWC = 510 * 64
  int t = threadIdx.x;
  // load: float4 across pixels (16B-aligned: HCWC%4==0, p0%64==0)
#pragma unroll
  for (int it = 0; it < 4; ++it) {
    int lin = it * 256 + t;
    int c = lin >> 4;            // 0..63
    int pq = (lin & 15) * 4;     // 0..60
    float4 v = *(const float4*)&M1[((size_t)b * 64 + c) * HCWC + p0 + pq];
    ld[pq + 0][c] = v.x; ld[pq + 1][c] = v.y; ld[pq + 2][c] = v.z; ld[pq + 3][c] = v.w;
  }
  __syncthreads();
  if (t < 64) {
    float s = 0.0f;
#pragma unroll
    for (int c = 0; c < 64; ++c) { float v = ld[t][c]; s += v * v; }
    sc[t] = 1.0f / fmaxf(sqrtf(s), 1e-12f);
  }
  __syncthreads();
  // store: float4 across channels (16B-aligned: base*64 + 4*cq)
#pragma unroll
  for (int it = 0; it < 4; ++it) {
    int lin = it * 256 + t;
    int p = lin >> 4;            // 0..63
    int cq = (lin & 15) * 4;     // 0..60
    float iv = sc[p];
    float4 v = make_float4(ld[p][cq + 0] * iv, ld[p][cq + 1] * iv,
                           ld[p][cq + 2] * iv, ld[p][cq + 3] * iv);
    *(float4*)&m1n[((size_t)b * HCWC + p0 + p) * 64 + cq] = v;
  }
}

// -------- 7. bicubic features + WMMA Gram-diagonal L2 normalization --------
__global__ void __launch_bounds__(512)
k_feats(const float* __restrict__ m1n, const int* __restrict__ topI,
        float* __restrict__ out) {
  __shared__ float sF[16][68];
  __shared__ float sInv[16];
  int b = blockIdx.x >> 8;             // 256 blocks per batch
  int kb = (blockIdx.x & 255) * 16;    // 16 keypoints per block
  int w = threadIdx.x >> 5, lane = threadIdx.x & 31;  // wave w handles keypoint kb+w

  int id = topI[b * TOPK + kb + w];
  float kx = (float)(id % W_), ky = (float)(id / W_);
  float ux = kx * (240.0f / 1919.0f) - 0.5f;
  float uy = ky * (136.0f / 1087.0f) - 0.5f;
  float xf = floorf(ux), yf = floorf(uy);
  float tx = ux - xf, ty = uy - yf;
  int x0 = (int)xf, y0 = (int)yf;
  float wx[4] = {cub2(tx + 1.0f), cub1(tx), cub1(1.0f - tx), cub2(2.0f - tx)};
  float wy[4] = {cub2(ty + 1.0f), cub1(ty), cub1(1.0f - ty), cub2(2.0f - ty)};

  float f0 = 0.0f, f1 = 0.0f;  // lane holds channels lane and lane+32
#pragma unroll
  for (int i = 0; i < 4; ++i) {
    int iy = y0 + i - 1;
    bool ry = (iy >= 0) && (iy < HC);
    if (ry && i < 3) {
      int nx = x0 - 1; nx = nx < 0 ? 0 : (nx > WC - 1 ? WC - 1 : nx);
      __builtin_prefetch(m1n + ((size_t)b * HCWC + (size_t)(iy + 1 < HC ? iy + 1 : iy) * WC + nx) * 64, 0, 0);
    }
#pragma unroll
    for (int j = 0; j < 4; ++j) {
      int ix = x0 + j - 1;
      if (ry && ix >= 0 && ix < WC) {  // zeros padding
        const float* p = m1n + ((size_t)b * HCWC + (size_t)iy * WC + ix) * 64;
        float wgt = wy[i] * wx[j];
        f0 = fmaf(wgt, p[lane], f0);
        f1 = fmaf(wgt, p[lane + 32], f1);
      }
    }
  }
  sF[w][lane] = f0;
  sF[w][lane + 32] = f1;
  __syncthreads();

  // Wave 0: squared norms of the 16x64 tile = diag(F * F^T) on the matrix pipe.
  if (threadIdx.x < 32) {
    int m = lane & 15;
    bool hi = lane >= 16;
    v8f acc = {0.f, 0.f, 0.f, 0.f, 0.f, 0.f, 0.f, 0.f};
#pragma unroll
    for (int ch = 0; ch < 64; ch += 32) {
      v16h a, bb;
#pragma unroll
      for (int e = 0; e < 16; ++e) {
        int v = e >> 1;  // A layout: 16-bit A-matrix 16x32 (ISA 7.12.2)
        int ka = (v < 4) ? (2 * v + (e & 1) + (hi ? 8 : 0))
                         : (16 + 2 * (v - 4) + (e & 1) + (hi ? 8 : 0));
        a[e] = (_Float16)sF[m][ch + ka];
        int kbk = (e < 8) ? (e + (hi ? 8 : 0)) : (16 + (e - 8) + (hi ? 8 : 0));  // B layout: same K order as A
        bb[e] = (_Float16)sF[m][ch + kbk];
      }
      acc = __builtin_amdgcn_wmma_f32_16x16x32_f16(false, a, false, bb,
                                                   (short)0, acc, false, false);
    }
    // D layout: VGPR r, lanes 0-15 -> M=r, lanes 16-31 -> M=r+8; N = lane%16
    float d = 0.0f;
#pragma unroll
    for (int r = 0; r < 8; ++r) {
      int row = r + (hi ? 8 : 0);
      if (row == m) d = acc[r];  // diagonal element (m,m)
    }
    bool owner = hi ? (m >= 8) : (m < 8);
    if (owner) sInv[m] = 1.0f / fmaxf(sqrtf(d), 1e-12f);
  }
  __syncthreads();

  float* fo = out + (size_t)B_ * TOPK * 3;  // after mkpts(2) + scores(1)
  for (int i = threadIdx.x; i < 16 * 64; i += 512) {
    int kk = i >> 6, c = i & 63;
    fo[((size_t)b * TOPK + kb + kk) * 64 + c] = sF[kk][c] * sInv[kk];
  }
}

// ---------------------------------------------------------------------------
extern "C" void kernel_launch(void* const* d_in, const int* in_sizes, int n_in,
                              void* d_out, int out_size, void* d_ws, size_t ws_size,
                              hipStream_t stream) {
  (void)in_sizes; (void)n_in; (void)out_size; (void)ws_size;
  const float* M1 = (const float*)d_in[0];
  const float* K1 = (const float*)d_in[1];
  const float* H1 = (const float*)d_in[2];
  float* out = (float*)d_out;

  // workspace layout (requires ~287 MB)
  float* k1hp   = (float*)d_ws;                        // B*HP*WP
  float* m1n    = k1hp + (size_t)B_ * HP * WP;         // B*HCWC*64 (NHWC, normalized)
  float* posS   = m1n + (size_t)B_ * HCWC * 64;        // B*MAXPOS
  int*   posI   = (int*)(posS + (size_t)B_ * MAXPOS);  // B*MAXPOS
  int*   posCnt = posI + (size_t)B_ * MAXPOS;          // B
  int*   hist   = posCnt + B_;                         // B*NBIN
  int*   cnts   = hist + B_ * NBIN;                    // 2 counters per batch
  int*   cutinfo= cnts + B_ * 2;                       // cutBin, nAbove
  float* candS  = (float*)(cutinfo + B_ * 2);          // B*TOPK
  int*   candI  = (int*)(candS + (size_t)B_ * TOPK);   // B*TOPK
  int*   topI   = candI + B_ * TOPK;                   // B*TOPK

  hipMemsetAsync(posCnt, 0, (size_t)B_ * sizeof(int), stream);
  hipMemsetAsync(hist,   0, (size_t)B_ * NBIN * sizeof(int), stream);
  hipMemsetAsync(cnts,   0, (size_t)B_ * 2 * sizeof(int), stream);
  hipMemsetAsync(candS,  0, (size_t)B_ * TOPK * sizeof(float), stream);
  hipMemsetAsync(candI,  0, (size_t)B_ * TOPK * sizeof(int), stream);

  k_zero_border<<<(B_ * 24128 + 255) / 256, 256, 0, stream>>>(k1hp);
  k_softmax_shuffle<<<(B_ * HCWC) / 256, 256, 0, stream>>>(K1, k1hp);
  k_nms_score<<<dim3(W_ / 32, H_ / 8, B_), 256, 0, stream>>>(k1hp, H1, posS, posI, posCnt, hist);
  k_cutoff<<<B_, 32, 0, stream>>>(hist, cutinfo);
  k_compact<<<(B_ * MAXPOS) / 256, 256, 0, stream>>>(posS, posI, posCnt, cutinfo, cnts, candS, candI);
  k_sort<<<B_, 512, 0, stream>>>(candS, candI, topI, out);
  k_norm_tr<<<B_ * 510, 256, 0, stream>>>(M1, m1n);
  k_feats<<<B_ * 256, 512, 0, stream>>>(m1n, topI, out);
}
